// PositionAwareWindowedAttention_52407190946370
// MI455X (gfx1250) — compile-verified, matrix operand order checked by reading, not run on hardware
//
#include <hip/hip_runtime.h>
#include <math.h>

// Problem constants (match reference)
#define B_     2
#define T_     4096
#define S_     4096
#define G_     64
#define D_     1024
#define H_     16
#define W_     512
#define CHUNK_ 256
#define HD_    64
#define NC_    (T_ / CHUNK_)
#define NKEY_  (G_ + W_)      // 576 keys per (b,h,chunk)
#define VT_STRIDE 584         // 576 + 8 pad, multiple of 8 elems -> 16B aligned rows

#if __has_builtin(__builtin_amdgcn_global_load_async_to_lds_b128)
#define HAVE_ASYNC_LDS 1
#endif

typedef __attribute__((ext_vector_type(16))) __bf16 v16bf;
typedef __attribute__((ext_vector_type(8)))  __bf16 v8bf;
typedef __attribute__((ext_vector_type(8)))  float  v8f;
typedef __attribute__((ext_vector_type(4)))  float  v4f;
typedef __attribute__((ext_vector_type(4)))  int    v4i;
typedef __attribute__((ext_vector_type(4)))  unsigned short u16x4;

// ---------- helpers ----------

__device__ __forceinline__ unsigned short f2bf_bits(float f) {
  union { float f; unsigned u; } a; a.f = f;
  unsigned u = a.u + 0x7fffu + ((a.u >> 16) & 1u);   // round-to-nearest-even
  return (unsigned short)(u >> 16);
}

__device__ __forceinline__ unsigned pack_bf16x2(float a, float b) {
  return (unsigned)f2bf_bits(a) | ((unsigned)f2bf_bits(b) << 16);
}

__device__ __forceinline__ v8f zero8() {
  v8f z = {0.f, 0.f, 0.f, 0.f, 0.f, 0.f, 0.f, 0.f};
  return z;
}

// Build a 16x32 bf16 A/B fragment half for this lane from bf16 memory.
// Elements [0..7] come from p[0..7], elements [8..15] from p[16..23].
__device__ __forceinline__ v16bf load_frag_bf16(const unsigned short* p) {
  v8bf lo = *(const v8bf*)(p);
  v8bf hi = *(const v8bf*)(p + 16);
  return __builtin_shufflevector(lo, hi, 0,1,2,3,4,5,6,7,8,9,10,11,12,13,14,15);
}

__device__ __forceinline__ v8f wmma_bf16(v16bf a, v16bf b, v8f c) {
  // D = A(16x32) x B(32x16) + C(16x16 f32)
  return __builtin_amdgcn_wmma_f32_16x16x32_bf16(false, a, false, b,
                                                 (short)0, c, false, false);
}

// all-ones 32x16 B fragment (bf16 1.0 = 0x3F80): P @ ones -> row sums in every column
__device__ __forceinline__ v16bf ones_frag() {
  union { unsigned short s; __bf16 b; } cv; cv.s = 0x3F80;
  v16bf o;
#pragma unroll
  for (int i = 0; i < 16; ++i) o[i] = cv.b;
  return o;
}

// max-reduce across the 16-lane half-wave using pure-VALU DPP butterfly
__device__ __forceinline__ float rowmax16(float x) {
  int t;
  t = __builtin_amdgcn_update_dpp(__float_as_int(x), __float_as_int(x),
                                  0xB1, 0xF, 0xF, false);      // quad_perm xor1
  x = fmaxf(x, __int_as_float(t));
  t = __builtin_amdgcn_update_dpp(__float_as_int(x), __float_as_int(x),
                                  0x4E, 0xF, 0xF, false);      // quad_perm xor2
  x = fmaxf(x, __int_as_float(t));
  t = __builtin_amdgcn_update_dpp(__float_as_int(x), __float_as_int(x),
                                  0x141, 0xF, 0xF, false);     // row_half_mirror
  x = fmaxf(x, __int_as_float(t));
  t = __builtin_amdgcn_update_dpp(__float_as_int(x), __float_as_int(x),
                                  0x140, 0xF, 0xF, false);     // row_mirror
  x = fmaxf(x, __int_as_float(t));
  return x;
}

// 16-byte global -> LDS stage: CDNA5 async DMA if available, else sync copy
__device__ __forceinline__ void stage16(const unsigned short* g, unsigned short* l) {
#ifdef HAVE_ASYNC_LDS
  unsigned short* gg = const_cast<unsigned short*>(g);
  __builtin_amdgcn_global_load_async_to_lds_b128(
      (__attribute__((address_space(1))) v4i*)gg,
      (__attribute__((address_space(3))) v4i*)l,
      0, 0);
#else
  *(uint4*)l = *(const uint4*)g;
#endif
}

__device__ __forceinline__ void wait_async0() {
#ifdef HAVE_ASYNC_LDS
#if __has_builtin(__builtin_amdgcn_s_wait_asynccnt)
  __builtin_amdgcn_s_wait_asynccnt(0);
#else
  asm volatile("s_wait_asynccnt 0" ::: "memory");
#endif
#endif
}

// ---------- kernel 0: f32 -> bf16 conversion, 4 elems/thread ----------

__global__ __launch_bounds__(256)
void cvt_f32_bf16_x4(const float* __restrict__ src, unsigned short* __restrict__ dst, int n4) {
  int i = blockIdx.x * 256 + threadIdx.x;
  if (i < n4) {
    v4f f = *(const v4f*)(src + (size_t)i * 4);
    u16x4 o;
#pragma unroll
    for (int j = 0; j < 4; ++j) o[j] = f2bf_bits(f[j]);
    *(u16x4*)(dst + (size_t)i * 4) = o;
  }
}

// ---------- kernel 1: GEMM  C = (A(bf16) @ W^T(bf16) + bias) * outScale ----------
// 256 threads = 8 waves; block tile 256x64; wave tile 32x64 (2 A frags share 4 B frags).

template <bool HEADOUT>
__global__ __launch_bounds__(256)
void gemm_bf16(const unsigned short* __restrict__ A,    // [Mtot][D] bf16
               const unsigned short* __restrict__ Wt,   // [D][D] bf16 row-major (N,K)
               const float* __restrict__ bias,          // [D]
               unsigned short* __restrict__ OutBf,      // used if HEADOUT
               float* __restrict__ OutF,                // used if !HEADOUT
               int Mb, int Mtot, float outScale) {
  const int tid  = threadIdx.x;
  const int wave = tid >> 5;
  const int lane = tid & 31;
  const int hi   = lane >> 4;
  const int ln   = lane & 15;
  const int koff = hi * 8;

  const int m0 = blockIdx.x * 256 + wave * 32;
  if (m0 >= Mtot) return;                    // wave-uniform: EXEC stays all-ones
  const int n0 = blockIdx.y * 64;

  const unsigned short* arow0 = A + (size_t)(m0 + ln) * D_ + koff;
  const unsigned short* arow1 = A + (size_t)(m0 + 16 + ln) * D_ + koff;

  v8f c[2][4];
#pragma unroll
  for (int r = 0; r < 2; ++r)
#pragma unroll
    for (int t = 0; t < 4; ++t) c[r][t] = zero8();

  for (int k = 0; k < D_; k += 32) {
    if (k + 32 < D_) {                        // prefetch next k-step
      __builtin_prefetch(arow0 + k + 32, 0, 3);
      __builtin_prefetch(arow1 + k + 32, 0, 3);
      __builtin_prefetch(Wt + (size_t)(n0 + ln) * D_ + k + 32 + koff, 0, 3);
    }
    v16bf a0 = load_frag_bf16(arow0 + k);
    v16bf a1 = load_frag_bf16(arow1 + k);
#pragma unroll
    for (int t = 0; t < 4; ++t) {
      v16bf b = load_frag_bf16(Wt + (size_t)(n0 + t * 16 + ln) * D_ + k + koff);
      c[0][t] = wmma_bf16(a0, b, c[0][t]);
      c[1][t] = wmma_bf16(a1, b, c[1][t]);
    }
  }

#pragma unroll
  for (int r = 0; r < 2; ++r) {
    const int gm0 = m0 + r * 16;             // 16-row sub-tile never crosses a batch
    if (HEADOUT) {
      const int bb    = gm0 / Mb;
      const int mbase = gm0 - bb * Mb;
#pragma unroll
      for (int t = 0; t < 4; ++t) {
        int   n  = n0 + t * 16 + ln;
        int   h  = n >> 6;
        int   hd = n & 63;
        float bs = bias[n];
#pragma unroll
        for (int v = 0; v < 8; ++v) {
          int m = mbase + v + 8 * hi;
          size_t off = (((size_t)bb * H_ + h) * (size_t)Mb + m) * HD_ + hd;
          OutBf[off] = f2bf_bits((c[r][t][v] + bs) * outScale);
        }
      }
    } else {
#pragma unroll
      for (int t = 0; t < 4; ++t) {
        int   n  = n0 + t * 16 + ln;
        float bs = bias[n];
#pragma unroll
        for (int v = 0; v < 8; ++v) {
          int gm = gm0 + v + 8 * hi;
          OutF[(size_t)gm * D_ + n] = c[r][t][v] + bs;
        }
      }
    }
  }
}

// ---------- kernel 2: flash attention over [G global + W windowed] keys ----------
// Grid: B*H*NC blocks, 256 threads = 8 waves; wave owns 32 queries (2 x 16-row tiles).
// K staged once into LDS via CDNA5 async-to-LDS DMA (each wave re-reads all 576 keys,
// so LDS staging removes 8x redundant global traffic and the global/window select).
// V staged transposed (sync; needs VGPR middle). Row-sum via P@ones WMMA; row-max
// via DPP; P stored packed with interleaved key columns.

__global__ __launch_bounds__(256)
void attn_kernel(const unsigned short* __restrict__ Qh,   // [B][H][T][HD] (pre-scaled)
                 const unsigned short* __restrict__ Kh,   // [B][H][S][HD]
                 const unsigned short* __restrict__ Vh,   // [B][H][S][HD]
                 const unsigned short* __restrict__ GKh,  // [B][H][G][HD]
                 const unsigned short* __restrict__ GVh,  // [B][H][G][HD]
                 const int* __restrict__ ratio,
                 unsigned short* __restrict__ AttnOut) {  // [B][T][D] bf16
  extern __shared__ char smem_raw[];
  unsigned short* Vt = (unsigned short*)smem_raw;                               // [HD][VT_STRIDE]
  unsigned short* Kl = (unsigned short*)(smem_raw + (size_t)HD_ * VT_STRIDE * 2);  // [NKEY][HD]
  unsigned short* Pb = Kl + (size_t)NKEY_ * HD_;                                // 8 x 32x32

  const int tid  = threadIdx.x;
  const int wave = tid >> 5;
  const int lane = tid & 31;
  const int hi   = lane >> 4;
  const int ln   = lane & 15;
  const int koff = hi * 8;

  const int blk = blockIdx.x;
  const int c   = blk % NC_;
  const int h   = (blk / NC_) % H_;
  const int b   = blk / (NC_ * H_);

  // window start (matches reference: trunc((c*CHUNK+CHUNK/2)/r), clamp)
  const int r0  = ratio[0];
  int center = (int)((float)(c * CHUNK_ + CHUNK_ / 2) / (float)r0);
  center = center < 0 ? 0 : (center > S_ - 1 ? S_ - 1 : center);
  int win = center - W_ / 2;
  win = win < 0 ? 0 : (win > S_ - W_ ? S_ - W_ : win);

  const size_t bh = (size_t)b * H_ + h;
  const unsigned short* Qbase  = Qh  + bh * T_ * HD_;
  const unsigned short* Kbase  = Kh  + bh * S_ * HD_;
  const unsigned short* Vbase  = Vh  + bh * S_ * HD_;
  const unsigned short* GKbase = GKh + bh * G_ * HD_;
  const unsigned short* GVbase = GVh + bh * G_ * HD_;

  // 1) Kick off K staging (row-major, async DMA path) ...
  for (int base = tid * 8; base < NKEY_ * HD_; base += 256 * 8) {
    int key = base >> 6;
    int d   = base & 63;                   // multiple of 8
    const unsigned short* srow = (key < G_)
        ? (GKbase + (size_t)key * HD_)
        : (Kbase + (size_t)(win + key - G_) * HD_);
    stage16(srow + d, Kl + (size_t)key * HD_ + d);
  }

  // 2) ... overlap with V transpose staging (interleaved key columns):
  // key = 32*kb + i  ->  col = 32*kb + ((i&15)<<1 | (i>>4))
  for (int base = tid * 8; base < NKEY_ * HD_; base += 256 * 8) {
    int key = base >> 6;
    int d   = base & 63;
    int i   = key & 31;
    int col = (key & ~31) | (((i & 15) << 1) | (i >> 4));
    const unsigned short* srow = (key < G_)
        ? (GVbase + (size_t)key * HD_)
        : (Vbase + (size_t)(win + key - G_) * HD_);
    union { uint4 u; unsigned short s[8]; } pk;
    pk.u = *(const uint4*)(srow + d);
#pragma unroll
    for (int j = 0; j < 8; ++j)
      Vt[(d + j) * VT_STRIDE + col] = pk.s[j];
  }
  wait_async0();
  __syncthreads();

  // Q fragments: 2 query tiles per wave
  const int q0 = c * CHUNK_ + wave * 32;
  v16bf aq[2][2];
#pragma unroll
  for (int r = 0; r < 2; ++r) {
    aq[r][0] = load_frag_bf16(Qbase + (size_t)(q0 + r * 16 + ln) * HD_ + 0  + koff);
    aq[r][1] = load_frag_bf16(Qbase + (size_t)(q0 + r * 16 + ln) * HD_ + 32 + koff);
  }

  float mrun[2][8], lrun[2][8];
  v8f oc[2][4];
#pragma unroll
  for (int r = 0; r < 2; ++r) {
#pragma unroll
    for (int v = 0; v < 8; ++v) { mrun[r][v] = -1e30f; lrun[r][v] = 0.f; }
#pragma unroll
    for (int t = 0; t < 4; ++t) oc[r][t] = zero8();
  }

  unsigned short* Pw = Pb + wave * (32 * 32);   // per-wave 32x32 bf16 scratch
  const v16bf onesf = ones_frag();

  for (int kb = 0; kb < NKEY_ / 32; ++kb) {
    const int j0 = kb * 32;

    // --- scores from LDS K: both query tiles share each K fragment pair ---
    v8f s[2][2];
#pragma unroll
    for (int t2 = 0; t2 < 2; ++t2) {
      const unsigned short* kp = Kl + (size_t)(j0 + t2 * 16 + ln) * HD_;
      v16bf bk0 = load_frag_bf16(kp + 0  + koff);
      v16bf bk1 = load_frag_bf16(kp + 32 + koff);
#pragma unroll
      for (int r = 0; r < 2; ++r) {
        v8f acc = zero8();
        acc = wmma_bf16(aq[r][0], bk0, acc);
        acc = wmma_bf16(aq[r][1], bk1, acc);
        s[r][t2] = acc;
      }
    }

    // --- online softmax: max via DPP butterfly; P stored packed (interleaved cols) ---
    float alpha[2][8];
#pragma unroll
    for (int r = 0; r < 2; ++r) {
#pragma unroll
      for (int v = 0; v < 8; ++v) {
        float s0 = s[r][0][v];
        float s1 = s[r][1][v];
        float cmax = rowmax16(fmaxf(s0, s1));
        float mnew = fmaxf(mrun[r][v], cmax);
        float p0 = __expf(s0 - mnew);
        float p1 = __expf(s1 - mnew);
        alpha[r][v] = __expf(mrun[r][v] - mnew);
        mrun[r][v]  = mnew;
        int prow = r * 16 + v + 8 * hi;
        // key j0+ln -> col 2*ln, key j0+16+ln -> col 2*ln+1: one b32 store
        *(unsigned*)(Pw + prow * 32 + 2 * ln) = pack_bf16x2(p0, p1);
      }
      // rescale running output and sums
#pragma unroll
      for (int t = 0; t < 4; ++t)
#pragma unroll
        for (int v = 0; v < 8; ++v) oc[r][t][v] *= alpha[r][v];
#pragma unroll
      for (int v = 0; v < 8; ++v) lrun[r][v] *= alpha[r][v];
    }

    // same-wave cross-lane LDS RAW: wait for all P stores before re-layout load
    asm volatile("s_wait_dscnt 0" ::: "memory");

    v16bf pf[2];
#pragma unroll
    for (int r = 0; r < 2; ++r)
      pf[r] = load_frag_bf16(Pw + (r * 16 + ln) * 32 + koff);

    // --- row sums via P @ ones: every column of C holds the row sum ---
#pragma unroll
    for (int r = 0; r < 2; ++r) {
      v8f rsum = wmma_bf16(pf[r], onesf, zero8());
#pragma unroll
      for (int v = 0; v < 8; ++v) lrun[r][v] += rsum[v];
    }

    // --- O(32x64) += P @ V: both query tiles share each V fragment ---
#pragma unroll
    for (int t = 0; t < 4; ++t) {
      const unsigned short* vp = Vt + (size_t)(t * 16 + ln) * VT_STRIDE + j0 + koff;
      v16bf bv = load_frag_bf16(vp);
#pragma unroll
      for (int r = 0; r < 2; ++r)
        oc[r][t] = wmma_bf16(pf[r], bv, oc[r][t]);
    }
  }

  // finalize: divide by row sums, store bf16 to [B][T][D] for the output GEMM
#pragma unroll
  for (int r = 0; r < 2; ++r) {
#pragma unroll
    for (int v = 0; v < 8; ++v) lrun[r][v] = 1.f / lrun[r][v];
#pragma unroll
    for (int t = 0; t < 4; ++t) {
      int dcol = t * 16 + ln;
#pragma unroll
      for (int v = 0; v < 8; ++v) {
        int q = q0 + r * 16 + v + 8 * hi;
        float o = oc[r][t][v] * lrun[r][v];
        AttnOut[((size_t)b * T_ + q) * D_ + h * HD_ + dcol] = f2bf_bits(o);
      }
    }
  }
}

// ---------- launcher ----------

extern "C" void kernel_launch(void* const* d_in, const int* in_sizes, int n_in,
                              void* d_out, int out_size, void* d_ws, size_t ws_size,
                              hipStream_t stream) {
  (void)in_sizes; (void)n_in; (void)out_size; (void)ws_size;

  const float* query = (const float*)d_in[0];
  const float* enc   = (const float*)d_in[1];
  const float* glob  = (const float*)d_in[2];
  const float* qw  = (const float*)d_in[3];  const float* qb  = (const float*)d_in[4];
  const float* kw  = (const float*)d_in[5];  const float* kb  = (const float*)d_in[6];
  const float* vw  = (const float*)d_in[7];  const float* vb  = (const float*)d_in[8];
  const float* gkw = (const float*)d_in[9];  const float* gkb = (const float*)d_in[10];
  const float* gvw = (const float*)d_in[11]; const float* gvb = (const float*)d_in[12];
  const float* ow  = (const float*)d_in[13]; const float* ob  = (const float*)d_in[14];
  const int*   ratio = (const int*)d_in[15];
  float* out = (float*)d_out;

  // workspace carve-up (bf16 as ushort)
  unsigned short* ws = (unsigned short*)d_ws;
  size_t off = 0;
  auto take = [&](size_t n) { unsigned short* p = ws + off; off += n; return p; };
  unsigned short* wq  = take((size_t)D_ * D_);
  unsigned short* wk  = take((size_t)D_ * D_);
  unsigned short* wv  = take((size_t)D_ * D_);
  unsigned short* wgk = take((size_t)D_ * D_);
  unsigned short* wgv = take((size_t)D_ * D_);
  unsigned short* wo  = take((size_t)D_ * D_);
  unsigned short* qbf = take((size_t)B_ * T_ * D_);   // bf16 activations
  unsigned short* ebf = take((size_t)B_ * S_ * D_);
  unsigned short* gbf = take((size_t)B_ * G_ * D_);
  unsigned short* qh  = take((size_t)B_ * T_ * D_);
  unsigned short* kh  = take((size_t)B_ * S_ * D_);
  unsigned short* vh  = take((size_t)B_ * S_ * D_);
  unsigned short* gkh = take((size_t)B_ * G_ * D_);
  unsigned short* gvh = take((size_t)B_ * G_ * D_);
  unsigned short* attn = take((size_t)B_ * T_ * D_);

  // 0) weights + activations -> bf16 (once; keeps cvt out of GEMM inner loops)
  const int nw4 = (D_ * D_) / 4;
  cvt_f32_bf16_x4<<<(nw4 + 255) / 256, 256, 0, stream>>>(qw,  wq,  nw4);
  cvt_f32_bf16_x4<<<(nw4 + 255) / 256, 256, 0, stream>>>(kw,  wk,  nw4);
  cvt_f32_bf16_x4<<<(nw4 + 255) / 256, 256, 0, stream>>>(vw,  wv,  nw4);
  cvt_f32_bf16_x4<<<(nw4 + 255) / 256, 256, 0, stream>>>(gkw, wgk, nw4);
  cvt_f32_bf16_x4<<<(nw4 + 255) / 256, 256, 0, stream>>>(gvw, wgv, nw4);
  cvt_f32_bf16_x4<<<(nw4 + 255) / 256, 256, 0, stream>>>(ow,  wo,  nw4);
  const int nq4 = (B_ * T_ * D_) / 4;
  const int ne4 = (B_ * S_ * D_) / 4;
  const int ng4 = (B_ * G_ * D_) / 4;
  cvt_f32_bf16_x4<<<(nq4 + 255) / 256, 256, 0, stream>>>(query, qbf, nq4);
  cvt_f32_bf16_x4<<<(ne4 + 255) / 256, 256, 0, stream>>>(enc,   ebf, ne4);
  cvt_f32_bf16_x4<<<(ng4 + 255) / 256, 256, 0, stream>>>(glob,  gbf, ng4);

  // 1) projections -> head-major bf16 (Q pre-scaled by 1/sqrt(HD) = 0.125)
  dim3 gq((B_ * T_ + 255) / 256, D_ / 64);     // 32 x 16
  gemm_bf16<true><<<gq, 256, 0, stream>>>(qbf, wq, qb, qh, nullptr, T_, B_ * T_, 0.125f);
  gemm_bf16<true><<<gq, 256, 0, stream>>>(ebf, wk, kb, kh, nullptr, S_, B_ * S_, 1.0f);
  gemm_bf16<true><<<gq, 256, 0, stream>>>(ebf, wv, vb, vh, nullptr, S_, B_ * S_, 1.0f);
  dim3 gg((B_ * G_ + 255) / 256, D_ / 64);     // 1 x 16 (waves beyond 128 rows early-out)
  gemm_bf16<true><<<gg, 256, 0, stream>>>(gbf, wgk, gkb, gkh, nullptr, G_, B_ * G_, 1.0f);
  gemm_bf16<true><<<gg, 256, 0, stream>>>(gbf, wgv, gvb, gvh, nullptr, G_, B_ * G_, 1.0f);

  // 2) attention (dynamic LDS: transposed V + LDS K + per-wave P scratch)
  size_t lds = (size_t)HD_ * VT_STRIDE * 2      // Vt       74,752 B
             + (size_t)NKEY_ * HD_ * 2          // K        73,728 B
             + (size_t)8 * 32 * 32 * 2;         // P        16,384 B  => 164,864 B
  attn_kernel<<<B_ * H_ * NC_, 256, lds, stream>>>(qh, kh, vh, gkh, gvh, ratio, attn);

  // 3) output projection -> f32
  gemm_bf16<false><<<gq, 256, 0, stream>>>(attn, wo, ob, nullptr, out, T_, B_ * T_, 1.0f);
}
